// GODEFunc_54709293416500
// MI455X (gfx1250) — matrix operand order repaired and checked by reading, not compiled, not run on hardware
//
#include <hip/hip_runtime.h>
#include <math.h>

#define BATCH 4
#define NN 8192
#define FF 64
#define KK 2
#define KC 64
#define ROWS 32
#define APITCH 68   // bf16 units per adj row in LDS (pad vs bank conflicts, keeps 8B align)
#define XPITCH 66   // bf16 units per x^T row in LDS
#define NCH (NN / KC)

typedef __attribute__((ext_vector_type(16))) __bf16  v16bf;
typedef __attribute__((ext_vector_type(8)))  float   v8f;
typedef __attribute__((ext_vector_type(2)))  float   v2f;

union FragBF { unsigned u[8]; v16bf v; };

__device__ __forceinline__ unsigned short f2bf(float f) {
  unsigned x = __float_as_uint(f);
  unsigned r = x + 0x7FFFu + ((x >> 16) & 1u);   // round-to-nearest-even
  return (unsigned short)(r >> 16);
}
__device__ __forceinline__ unsigned pack2(float lo, float hi) {
  return (unsigned)f2bf(lo) | ((unsigned)f2bf(hi) << 16);
}
__device__ __forceinline__ float sigmoidf_(float v) { return 1.0f / (1.0f + __expf(-v)); }

// W = (w * clip(d,0,1)) @ w^T   -> 64x64 into workspace
__global__ void gode_prep_w(const float* __restrict__ w, const float* __restrict__ d,
                            float* __restrict__ W) {
  int i = blockIdx.x;      // row
  int f = threadIdx.x;     // col
  float s = 0.f;
  for (int j = 0; j < FF; ++j) {
    float dj = d[j];
    dj = dj < 0.f ? 0.f : (dj > 1.f ? 1.f : dj);
    s += w[i * FF + j] * dj * w[f * FF + j];
  }
  W[i * FF + f] = s;
}

__global__ __launch_bounds__(256)
void gode_main(const float* __restrict__ x, const float* __restrict__ x0,
               const float* __restrict__ adj, const float* __restrict__ alpha,
               const float* __restrict__ beta, const float* __restrict__ Wmat,
               const float* __restrict__ conv_w, const float* __restrict__ conv_b,
               float* __restrict__ out) {
  __shared__ unsigned short sA[2][KK][ROWS][APITCH];     // adj tiles, bf16 row-major
  __shared__ unsigned short sX[2][BATCH][FF][XPITCH];    // x tiles, bf16 transposed [f][m]
  __shared__ float sXa[BATCH][ROWS][FF + 1];             // k-reduction exchange

  const int tid = threadIdx.x;
  const int lane = tid & 31;
  const int wave = tid >> 5;
  const int laneM = lane & 15;
  const int laneHi = lane >> 4;
  const int b = wave >> 1;      // batch owned by this wave
  const int kk = wave & 1;      // adj stack owned by this wave
  const int n0 = blockIdx.x * ROWS;

  // --- per-thread staging decode (constant across chunks) ---
  int xb[8], xm[8], xf[8];
#pragma unroll
  for (int i = 0; i < 8; ++i) {
    int u = tid + 256 * i;            // 2048 units: b(4) x m-pair(32) x f-quad(16)
    xb[i] = u >> 9;
    xm[i] = ((u >> 4) & 31) * 2;
    xf[i] = (u & 15) * 4;
  }
  int ak_[4], ar_[4], aq_[4];
#pragma unroll
  for (int i = 0; i < 4; ++i) {
    int u = tid + 256 * i;            // 1024 units: k(2) x row(32) x m-quad(16)
    ak_[i] = u >> 9;
    ar_[i] = (u >> 4) & 31;
    aq_[i] = (u & 15) * 4;
  }

  v8f acc[2][4];                      // [rowhalf][f-tile] 16x16 f32 accumulators
#pragma unroll
  for (int rh = 0; rh < 2; ++rh)
#pragma unroll
    for (int ft = 0; ft < 4; ++ft)
#pragma unroll
      for (int e = 0; e < 8; ++e) acc[rh][ft][e] = 0.f;

  float4 px[8][2];                    // prefetch regs: x rows m, m+1 (4 f's each)
  float4 pa[4];                       // prefetch regs: adj quads

  auto stage_load = [&](int m0) {
#pragma unroll
    for (int i = 0; i < 8; ++i) {
      size_t base = ((size_t)xb[i] * NN + (size_t)(m0 + xm[i])) * FF + xf[i];
      px[i][0] = *(const float4*)(x + base);
      px[i][1] = *(const float4*)(x + base + FF);
    }
#pragma unroll
    for (int i = 0; i < 4; ++i) {
      size_t base = ((size_t)ak_[i] * NN + (size_t)(n0 + ar_[i])) * (size_t)NN + (m0 + aq_[i]);
      pa[i] = *(const float4*)(adj + base);
    }
  };
  auto stage_store = [&](int buf) {
#pragma unroll
    for (int i = 0; i < 8; ++i) {
      int m = xm[i];
      unsigned short (*sxp)[XPITCH] = sX[buf][xb[i]];
      *(unsigned*)&sxp[xf[i] + 0][m] = pack2(px[i][0].x, px[i][1].x);
      *(unsigned*)&sxp[xf[i] + 1][m] = pack2(px[i][0].y, px[i][1].y);
      *(unsigned*)&sxp[xf[i] + 2][m] = pack2(px[i][0].z, px[i][1].z);
      *(unsigned*)&sxp[xf[i] + 3][m] = pack2(px[i][0].w, px[i][1].w);
    }
#pragma unroll
    for (int i = 0; i < 4; ++i) {
      unsigned* dst = (unsigned*)&sA[buf][ak_[i]][ar_[i]][aq_[i]];
      dst[0] = pack2(pa[i].x, pa[i].y);
      dst[1] = pack2(pa[i].z, pa[i].w);
    }
  };

  // --- main GEMM loop: acc[rh][ft] += adj_k[n-tile, m-chunk] @ x_b[m-chunk, f-tile] ---
  stage_load(0);
  stage_store(0);
  __syncthreads();

  for (int c = 0; c < NCH; ++c) {
    if (c + 1 < NCH) stage_load((c + 1) * KC);
    const int buf = c & 1;
#pragma unroll
    for (int ks = 0; ks < 2; ++ks) {            // two K=32 steps inside the chunk
      FragBF a0, a1;
#pragma unroll
      for (int v = 0; v < 8; ++v) {             // 16-bit A 16x32 layout
        int Kb = 32 * ks + ((v < 4) ? (2 * v + 8 * laneHi) : (16 + 2 * (v - 4) + 8 * laneHi));
        a0.u[v] = *(const unsigned*)&sA[buf][kk][laneM][Kb];
        a1.u[v] = *(const unsigned*)&sA[buf][kk][16 + laneM][Kb];
      }
#pragma unroll
      for (int ft = 0; ft < 4; ++ft) {
        FragBF bf;
        int f = 16 * ft + laneM;
#pragma unroll
        for (int r = 0; r < 8; ++r) {           // 16-bit B 32x16 layout
          int Kb = 32 * ks + ((r < 4) ? (2 * r + 8 * laneHi) : (16 + 2 * (r - 4) + 8 * laneHi));
          bf.u[r] = *(const unsigned*)&sX[buf][b][f][Kb];
        }
        acc[0][ft] = __builtin_amdgcn_wmma_f32_16x16x32_bf16(
            false, a0.v, false, bf.v, (short)0, acc[0][ft], false, false);
        acc[1][ft] = __builtin_amdgcn_wmma_f32_16x16x32_bf16(
            false, a1.v, false, bf.v, (short)0, acc[1][ft], false, false);
      }
    }
    if (c + 1 < NCH) stage_store((c + 1) & 1);
    __syncthreads();
  }

  // --- k-reduction across the (b,0)/(b,1) wave pair via LDS ---
  const float cw0 = conv_w[0], cw1 = conv_w[1], cb = conv_b[0];
  const float cwk = kk ? cw1 : cw0;
  const int myRH = kk, otherRH = 1 - kk;        // each wave finalizes 16 rows

#pragma unroll
  for (int ft = 0; ft < 4; ++ft)
#pragma unroll
    for (int r = 0; r < 8; ++r) {
      int row = 16 * otherRH + r + 8 * laneHi;
      sXa[b][row][16 * ft + laneM] = cwk * acc[otherRH][ft][r];
    }
  __syncthreads();

  float xa[4][8];
#pragma unroll
  for (int ft = 0; ft < 4; ++ft)
#pragma unroll
    for (int r = 0; r < 8; ++r) {
      int row = 16 * myRH + r + 8 * laneHi;
      xa[ft][r] = cwk * acc[myRH][ft][r] + sXa[b][row][16 * ft + laneM] + cb;
    }

  // --- xw = x @ W via f32 WMMA (keeps full fp32 precision on the identity-like path) ---
  v8f xw[4];
#pragma unroll
  for (int ft = 0; ft < 4; ++ft)
#pragma unroll
    for (int e = 0; e < 8; ++e) xw[ft][e] = 0.f;

  const size_t xrow = ((size_t)b * NN + (n0 + 16 * myRH + laneM)) * FF;
#pragma unroll 4
  for (int kc = 0; kc < 16; ++kc) {
    v2f ax;                                     // f32 A 16x4: K = v + 2*laneHi
    ax[0] = x[xrow + 4 * kc + 0 + 2 * laneHi];
    ax[1] = x[xrow + 4 * kc + 1 + 2 * laneHi];
#pragma unroll
    for (int ft = 0; ft < 4; ++ft) {
      v2f bw;                                   // f32 B 4x16: K = v + 2*laneHi, N = laneM
      bw[0] = Wmat[(4 * kc + 0 + 2 * laneHi) * FF + 16 * ft + laneM];
      bw[1] = Wmat[(4 * kc + 1 + 2 * laneHi) * FF + 16 * ft + laneM];
      xw[ft] = __builtin_amdgcn_wmma_f32_16x16x4_f32(
          false, ax, false, bw, (short)0, xw[ft], false, false);
    }
  }

  // --- final gates + tanh + store ---
#pragma unroll
  for (int r = 0; r < 8; ++r) {
    int nrow = n0 + 16 * myRH + r + 8 * laneHi;
    float ag = sigmoidf_(alpha[nrow]) * 0.5f;
    float bg = sigmoidf_(beta[nrow]);
#pragma unroll
    for (int ft = 0; ft < 4; ++ft) {
      int f = 16 * ft + laneM;
      size_t idx = ((size_t)b * NN + nrow) * FF + f;
      float xv = x[idx];
      float x0v = x0[idx];
      float pre = ag * xa[ft][r] - 2.0f * xv + xw[ft][r] + x0v * bg;
      out[idx] = tanhf(pre);
    }
  }
}

extern "C" void kernel_launch(void* const* d_in, const int* in_sizes, int n_in,
                              void* d_out, int out_size, void* d_ws, size_t ws_size,
                              hipStream_t stream) {
  const float* x      = (const float*)d_in[0];
  const float* x0     = (const float*)d_in[1];
  const float* adj    = (const float*)d_in[2];
  const float* alpha  = (const float*)d_in[3];
  const float* beta   = (const float*)d_in[4];
  const float* w      = (const float*)d_in[5];
  const float* d      = (const float*)d_in[6];
  const float* conv_w = (const float*)d_in[7];
  const float* conv_b = (const float*)d_in[8];
  float* out = (float*)d_out;
  float* W   = (float*)d_ws;                 // 64*64 f32 = 16 KB scratch

  gode_prep_w<<<FF, FF, 0, stream>>>(w, d, W);
  gode_main<<<NN / ROWS, 256, 0, stream>>>(x, x0, adj, alpha, beta, W,
                                           conv_w, conv_b, out);
}
